// MultiHeadAttention_38465727103460
// MI455X (gfx1250) — matrix operand order repaired and checked by reading, not compiled, not run on hardware
//
#include <hip/hip_runtime.h>
#include <hip/hip_bf16.h>

// ---- problem constants (from reference) ----
#define Bc 2
#define Sc 2048
#define Dc 1024
#define Hc 16
#define DHc 64
// All projection GEMMs: M = Bc*Sc = 4096, N = K = Dc = 1024

typedef __attribute__((ext_vector_type(16))) _Float16 v16h;
typedef __attribute__((ext_vector_type(8)))  _Float16 v8h;
typedef __attribute__((ext_vector_type(4)))  _Float16 v4h;
typedef __attribute__((ext_vector_type(8)))  float    v8f;
typedef __attribute__((ext_vector_type(4)))  float    v4f;

__device__ __forceinline__ v16h pack16(v8h lo, v8h hi) {
  v16h r;
#pragma unroll
  for (int i = 0; i < 8; ++i) { r[i] = lo[i]; r[8 + i] = hi[i]; }
  return r;
}

// async copy: 16 bytes global -> LDS, tracked by ASYNCcnt (CDNA5 path)
__device__ __forceinline__ void async_g2l_b128(const _Float16* g, unsigned lds_byte_off) {
  asm volatile("global_load_async_to_lds_b128 %0, %1, off"
               :: "v"(lds_byte_off), "v"(g) : "memory");
}
__device__ __forceinline__ void wait_asynccnt0() {
  asm volatile("s_wait_asynccnt 0x0" ::: "memory");
}

// ---------------------------------------------------------------------------
// weight f32 -> f16 pre-conversion (done once; weights are reused 32x per GEMM)
// ---------------------------------------------------------------------------
__global__ __launch_bounds__(256) void cvt_w_kernel(const float* __restrict__ in,
                                                    _Float16* __restrict__ out) {
  int idx = (blockIdx.x * 256 + threadIdx.x) * 4;  // Dc*Dc is an exact multiple
  v4f v = *(const v4f*)(in + idx);
  v4h h;
#pragma unroll
  for (int i = 0; i < 4; ++i) h[i] = (_Float16)v[i];
  *(v4h*)(out + idx) = h;
}

// ---------------------------------------------------------------------------
// GEMM: Out[4096, 1024] = A[4096,1024] @ W16[1024,1024]^T + bias
// Cooperative 8-wave workgroup computes a 128(M) x 128(N) tile.
// The W k-slice (128 x 32 f16 = 8 KB) is staged in LDS with async
// global->LDS copies, double-buffered (16 KB), so all 8 waves share one
// fetch of B while per-wave A fragments stream from global.
// A16:   A is f16 (attention output) vs f32 (x inputs)
// OMODE: 0 = f16 row-major [M,N] ; 1 = f16 V-transposed [B,H,DH,S] ; 2 = f32
// ---------------------------------------------------------------------------
template <bool A16, int OMODE>
__global__ __launch_bounds__(256) void gemm_mha(const void* __restrict__ Ain,
                                                const _Float16* __restrict__ W16,
                                                const float* __restrict__ bias,
                                                void* __restrict__ Out) {
  extern __shared__ _Float16 bsh[];  // 2 x (128 rows x 32 halves) = 16 KB

  const int tid   = threadIdx.x;
  const int wave  = tid >> 5;
  const int lane  = tid & 31;
  const int lm    = lane & 15;
  const int base  = (lane < 16) ? 0 : 8;   // A-fragment K sub-offset
  const int koffB = (lane < 16) ? 0 : 8;   // B-fragment K sub-offset

  const int tileM = blockIdx.x;            // 0..31
  const int n0    = blockIdx.y * 128;      // N tile origin
  const int m_row = tileM * 128 + wave * 16 + lm;

  // this thread's two async 16B transfers per k-step (512 x 16B = 8 KB slice)
  const int r0 = tid >> 2, c0 = tid & 3;           // rows 0..63
  const int r1 = r0 + 64,  c1 = c0;                // rows 64..127
  const _Float16* g0 = W16 + (size_t)(n0 + r0) * Dc + c0 * 8;
  const _Float16* g1 = W16 + (size_t)(n0 + r1) * Dc + c1 * 8;
  const unsigned  l0 = (unsigned)(r0 * 64 + c0 * 16);  // byte offset in buffer
  const unsigned  l1 = (unsigned)(r1 * 64 + c1 * 16);

  v8f acc[8];
#pragma unroll
  for (int j = 0; j < 8; ++j) acc[j] = {};

  // prefill buffer 0 with k-slice 0
  async_g2l_b128(g0, l0);
  async_g2l_b128(g1, l1);

  for (int kt = 0; kt < Dc / 32; ++kt) {
    wait_asynccnt0();    // my async writes into buf[kt&1] done
    __syncthreads();     // everyone's are done
    if (kt + 1 < Dc / 32) {
      const int kn = (kt + 1) * 32;
      const unsigned bo = (unsigned)(((kt + 1) & 1) * 8192);
      async_g2l_b128(g0 + kn, l0 + bo);
      async_g2l_b128(g1 + kn, l1 + bo);
    }

    // A fragment for k = kt*32 .. +31
    const int kk = kt * 32;
    v16h afrag;
    if (A16) {
      const _Float16* ap = (const _Float16*)Ain + (size_t)m_row * Dc + kk + base;
      afrag = pack16(*(const v8h*)ap, *(const v8h*)(ap + 16));
    } else {
      const float* ap = (const float*)Ain + (size_t)m_row * Dc + kk + base;
      v4f a0 = *(const v4f*)(ap);
      v4f a1 = *(const v4f*)(ap + 4);
      v4f a2 = *(const v4f*)(ap + 16);
      v4f a3 = *(const v4f*)(ap + 20);
#pragma unroll
      for (int i = 0; i < 4; ++i) {
        afrag[i]      = (_Float16)a0[i];
        afrag[4 + i]  = (_Float16)a1[i];
        afrag[8 + i]  = (_Float16)a2[i];
        afrag[12 + i] = (_Float16)a3[i];
      }
      if (kk + 32 < Dc) __builtin_prefetch(ap + 32, 0, 0);
    }

    // 8 B-fragments from the shared LDS slice
    const _Float16* bufp = bsh + ((kt & 1) ? 4096 : 0);
#pragma unroll
    for (int j = 0; j < 8; ++j) {
      const _Float16* bp = bufp + (j * 16 + lm) * 32 + koffB;
      v16h bfrag = pack16(*(const v8h*)bp, *(const v8h*)(bp + 16));
      acc[j] = __builtin_amdgcn_wmma_f32_16x16x32_f16(
          false, afrag, false, bfrag, (short)0, acc[j], false, false);
    }
  }

#pragma unroll
  for (int j = 0; j < 8; ++j) {
    const int n  = n0 + j * 16 + lm;
    const float bv = bias[n];
#pragma unroll
    for (int r = 0; r < 8; ++r) {
      const int m   = tileM * 128 + wave * 16 + ((lane < 16) ? r : r + 8);
      const float v = acc[j][r] + bv;
      if (OMODE == 0) {
        ((_Float16*)Out)[(size_t)m * Dc + n] = (_Float16)v;
      } else if (OMODE == 2) {
        ((float*)Out)[(size_t)m * Dc + n] = v;
      } else {  // V-transposed f16: [B,H,DH,S]
        const int b = m >> 11, s = m & (Sc - 1);
        const int h = n >> 6,  dh = n & (DHc - 1);
        ((_Float16*)Out)[(((size_t)(b * Hc + h) * DHc + dh) * Sc) + s] = (_Float16)v;
      }
    }
  }
}

// ---------------------------------------------------------------------------
// attention: one wave per (b, h, 16-query tile)
//   phase 1: scores[16,2048] = (Q tile @ K^T) * 1/8 + mask  -> LDS (f32, 128 KB)
//   softmax stats: per-row max & sumexp, halves combined via shfl_xor(16)
//   phase 2: O[16,64] = softmax(scores) @ V  (A-fragments built with exp on the fly)
// ---------------------------------------------------------------------------
__global__ __launch_bounds__(32) void attn_kernel(const _Float16* __restrict__ Qh,
                                                  const _Float16* __restrict__ Kh,
                                                  const _Float16* __restrict__ Vt,
                                                  const unsigned char* __restrict__ mask,
                                                  _Float16* __restrict__ O) {
  extern __shared__ float sc[];  // [16][Sc] scores

  const int lane  = threadIdx.x & 31;
  const int lm    = lane & 15;
  const int base  = (lane < 16) ? 0 : 8;
  const int koffB = (lane < 16) ? 0 : 8;
  const int q0 = blockIdx.x * 16;
  const int h  = blockIdx.y;
  const int b  = blockIdx.z;

  // Q tile A-fragments: d = 0..63 in two K-steps of 32
  const _Float16* qp = Qh + (size_t)(b * Sc + q0 + lm) * Dc + h * DHc;
  v16h aq[2];
#pragma unroll
  for (int ks = 0; ks < 2; ++ks) {
    const _Float16* p = qp + ks * 32 + base;
    aq[ks] = pack16(*(const v8h*)p, *(const v8h*)(p + 16));
  }

  // ---- phase 1: scores -> LDS ----
  for (int kt = 0; kt < Sc / 16; ++kt) {
    v8f acc = {};
    const _Float16* kp = Kh + (size_t)(b * Sc + kt * 16 + lm) * Dc + h * DHc + koffB;
#pragma unroll
    for (int ks = 0; ks < 2; ++ks) {
      v16h bf = pack16(*(const v8h*)(kp + ks * 32), *(const v8h*)(kp + ks * 32 + 16));
      acc = __builtin_amdgcn_wmma_f32_16x16x32_f16(
          false, aq[ks], false, bf, (short)0, acc, false, false);
    }
    const int kkn = kt * 16 + lm;
    const float madd = mask[(size_t)b * Sc + kkn] ? 0.0f : -1e9f;
#pragma unroll
    for (int r = 0; r < 8; ++r) {
      const int m = (lane < 16) ? r : r + 8;
      sc[m * Sc + kkn] = acc[r] * 0.125f + madd;
    }
  }
  __syncthreads();

  // ---- softmax stats: lane handles row lm, half (lane>=16) of columns ----
  const float* rp = sc + lm * Sc + ((lane < 16) ? 0 : 1024);
  float mx = -3.0e38f;
  for (int c = 0; c < 1024; c += 4) {
    v4f t = *(const v4f*)(rp + c);
    mx = fmaxf(mx, fmaxf(fmaxf(t[0], t[1]), fmaxf(t[2], t[3])));
  }
  mx = fmaxf(mx, __shfl_xor(mx, 16, 32));
  float sum = 0.0f;
  for (int c = 0; c < 1024; c += 4) {
    v4f t = *(const v4f*)(rp + c);
    sum += __expf(t[0] - mx) + __expf(t[1] - mx) + __expf(t[2] - mx) + __expf(t[3] - mx);
  }
  sum += __shfl_xor(sum, 16, 32);
  const float inv = 1.0f / sum;
  __syncthreads();

  // ---- phase 2: O = P @ V ----
  v8f oacc[4];
#pragma unroll
  for (int j = 0; j < 4; ++j) oacc[j] = {};

  const _Float16* vbase = Vt + (size_t)(b * Hc + h) * DHc * Sc;
  for (int kt = 0; kt < Sc / 32; ++kt) {
    // A-fragment: probabilities for row lm, K = kt*32 + {base..+7, base+16..+23}
    const float* pr = sc + lm * Sc + kt * 32 + base;
    v4f p0 = *(const v4f*)(pr);
    v4f p1 = *(const v4f*)(pr + 4);
    v4f p2 = *(const v4f*)(pr + 16);
    v4f p3 = *(const v4f*)(pr + 20);
    v16h af;
#pragma unroll
    for (int i = 0; i < 4; ++i) {
      af[i]      = (_Float16)(__expf(p0[i] - mx) * inv);
      af[4 + i]  = (_Float16)(__expf(p1[i] - mx) * inv);
      af[8 + i]  = (_Float16)(__expf(p2[i] - mx) * inv);
      af[12 + i] = (_Float16)(__expf(p3[i] - mx) * inv);
    }
#pragma unroll
    for (int j = 0; j < 4; ++j) {
      const _Float16* vp = vbase + (size_t)(j * 16 + lm) * Sc + kt * 32 + koffB;
      v16h bf = pack16(*(const v8h*)vp, *(const v8h*)(vp + 16));
      oacc[j] = __builtin_amdgcn_wmma_f32_16x16x32_f16(
          false, af, false, bf, (short)0, oacc[j], false, false);
    }
  }

  // store attention output (pre-Wo) as f16 [B,S,D]
#pragma unroll
  for (int j = 0; j < 4; ++j) {
    const int d = j * 16 + lm;
#pragma unroll
    for (int r = 0; r < 8; ++r) {
      const int m = (lane < 16) ? r : r + 8;
      O[(size_t)(b * Sc + q0 + m) * Dc + h * DHc + d] = (_Float16)oacc[j][r];
    }
  }
}

// ---------------------------------------------------------------------------
extern "C" void kernel_launch(void* const* d_in, const int* in_sizes, int n_in,
                              void* d_out, int out_size, void* d_ws, size_t ws_size,
                              hipStream_t stream) {
  const float* x_q = (const float*)d_in[0];
  const float* x_k = (const float*)d_in[1];
  const float* x_v = (const float*)d_in[2];
  const unsigned char* mask = (const unsigned char*)d_in[3];
  const float* Wq = (const float*)d_in[4];
  const float* bq = (const float*)d_in[5];
  const float* Wk = (const float*)d_in[6];
  const float* bk = (const float*)d_in[7];
  const float* Wv = (const float*)d_in[8];
  const float* bv = (const float*)d_in[9];
  const float* Wo = (const float*)d_in[10];
  const float* bo = (const float*)d_in[11];

  // workspace carving (f16 elements)
  const size_t NE  = (size_t)Bc * Sc * Dc;  // 4,194,304
  const size_t NW  = (size_t)Dc * Dc;       // 1,048,576
  _Float16* Qh   = (_Float16*)d_ws;
  _Float16* Kh   = Qh + NE;
  _Float16* Vt   = Kh + NE;   // [B,H,DH,S]
  _Float16* Oatt = Vt + NE;
  _Float16* Wq16 = Oatt + NE;
  _Float16* Wk16 = Wq16 + NW;
  _Float16* Wv16 = Wk16 + NW;
  _Float16* Wo16 = Wv16 + NW;

  // 1) convert weights to f16 once
  {
    dim3 g((unsigned)(NW / (256 * 4))), t(256);
    cvt_w_kernel<<<g, t, 0, stream>>>(Wq, Wq16);
    cvt_w_kernel<<<g, t, 0, stream>>>(Wk, Wk16);
    cvt_w_kernel<<<g, t, 0, stream>>>(Wv, Wv16);
    cvt_w_kernel<<<g, t, 0, stream>>>(Wo, Wo16);
  }

  // 2) Q/K/V projections (V written transposed per-head)
  {
    dim3 g(32, 8), t(256);
    size_t lds = 16384;  // double-buffered 8 KB B slices
    gemm_mha<false, 0><<<g, t, lds, stream>>>(x_q, Wq16, bq, Qh);
    gemm_mha<false, 0><<<g, t, lds, stream>>>(x_k, Wk16, bk, Kh);
    gemm_mha<false, 1><<<g, t, lds, stream>>>(x_v, Wv16, bv, Vt);
  }

  // 3) attention (one wave per (b,h,q-tile); 128 KB dynamic LDS for scores)
  {
    dim3 g(Sc / 16, Hc, Bc), t(32);
    size_t lds = (size_t)16 * Sc * sizeof(float);
    attn_kernel<<<g, t, lds, stream>>>(Qh, Kh, Vt, mask, Oatt);
  }

  // 4) output projection -> f32 d_out
  {
    dim3 g(32, 8), t(256);
    size_t lds = 16384;
    gemm_mha<true, 2><<<g, t, lds, stream>>>(Oatt, Wo16, bo, (float*)d_out);
  }
}
